// VQAEMG_28278064677185
// MI455X (gfx1250) — matrix operand order repaired and checked by reading, not compiled, not run on hardware
//
#include <hip/hip_runtime.h>
#include <hip/hip_bf16.h>
#include <math.h>

typedef __attribute__((ext_vector_type(16))) _Float16 v16h;
typedef __attribute__((ext_vector_type(8)))  _Float16 v8h;
typedef __attribute__((ext_vector_type(8)))  float    v8f;

#define M_  16384   // B*N tokens
#define E_  768
#define D_  256
#define K_  8192

// ---------------- WMMA fragment helpers (16x16x32 f16, wave32) ----------------
// A 16x32 (MxK), row-major source. lane<16: row=lane, K {0..7,16..23};
// lane>=16: row=lane-16, K {8..15,24..31}.
__device__ __forceinline__ v16h load_frag_a(const _Float16* __restrict__ A, int lda,
                                            int k0, int lane) {
    const _Float16* p = A + (size_t)(lane & 15) * lda + k0 + ((lane >> 4) << 3);
    v8h lo = *(const v8h*)(p);
    v8h hi = *(const v8h*)(p + 16);
    v16h r;
#pragma unroll
    for (int i = 0; i < 8; ++i) { r[i] = lo[i]; r[i + 8] = hi[i]; }
    return r;
}

// B 32x16 (KxN) from N-major (transposed) source BT[n][k]: lane holds column
// n = lane&15, K {0..7,16..23} (lane<16) or {8..15,24..31} (lane>=16).
__device__ __forceinline__ v16h load_frag_b(const _Float16* __restrict__ BT, int ldb,
                                            int k0, int lane) {
    const _Float16* p = BT + (size_t)(lane & 15) * ldb + k0 + ((lane >> 4) << 3);
    v8h lo = *(const v8h*)(p);
    v8h hi = *(const v8h*)(p + 16);
    v16h r;
#pragma unroll
    for (int i = 0; i < 8; ++i) { r[i] = lo[i]; r[i + 8] = hi[i]; }
    return r;
}

// ---------------- conversion / transpose kernels ----------------
__global__ void k_cvt_f16(const float* __restrict__ src, _Float16* __restrict__ dst, int n) {
    for (int i = blockIdx.x * blockDim.x + threadIdx.x; i < n; i += gridDim.x * blockDim.x)
        dst[i] = (_Float16)src[i];
}

// src fp32 [Krows][Ncols] -> dst f16 [Ncols][Krows]
__global__ void k_transpose_f16(const float* __restrict__ src, _Float16* __restrict__ dst,
                                int Krows, int Ncols) {
    int n = Krows * Ncols;
    for (int i = blockIdx.x * blockDim.x + threadIdx.x; i < n; i += gridDim.x * blockDim.x) {
        int k = i / Ncols, c = i % Ncols;
        dst[(size_t)c * Krows + k] = (_Float16)src[i];
    }
}

// codebook fp32 [K_][D_] -> f16 copy + fp32 row squared norms. One wave per row.
__global__ void k_cbprep(const float* __restrict__ cb, _Float16* __restrict__ cb16,
                         float* __restrict__ cbn) {
    int lane = threadIdx.x & 31, wave = threadIdx.x >> 5;
    int row = blockIdx.x * 8 + wave;
    const float* p = cb + (size_t)row * D_;
    float s = 0.f;
#pragma unroll
    for (int i = 0; i < 8; ++i) {
        float v = p[i * 32 + lane];
        cb16[(size_t)row * D_ + i * 32 + lane] = (_Float16)v;
        s += v * v;
    }
#pragma unroll
    for (int off = 16; off > 0; off >>= 1) s += __shfl_xor(s, off, 32);
    if (lane == 0) cbn[row] = s;
}

// ---------------- tiled WMMA GEMM: C[M,N] = act(A[M,K] @ BT[N,K]^T + bias) ----
// block = 256 threads = 8 waves, M_block = 128, N_block = 64 (4 n-tiles/wave).
template<int KDIM, bool TANH>
__global__ void k_gemm(const _Float16* __restrict__ A, const _Float16* __restrict__ BT,
                       const float* __restrict__ bias, _Float16* __restrict__ Ch,
                       float* __restrict__ Cf, int NDIM) {
    __shared__ __align__(16) _Float16 Bs[64 * 32];
    int tid = threadIdx.x, lane = tid & 31, wave = tid >> 5;
    int m0 = blockIdx.x * 128 + wave * 16;
    int n0 = blockIdx.y * 64;
    const _Float16* Arow = A + (size_t)m0 * KDIM;
    v8f acc[4] = {};
    for (int k0 = 0; k0 < KDIM; k0 += 32) {
        __syncthreads();
        int r = tid >> 2, s = tid & 3;   // 64 rows x 32 halves, 16B per thread
        *(v8h*)&Bs[r * 32 + s * 8] =
            *(const v8h*)&BT[(size_t)(n0 + r) * KDIM + k0 + s * 8];
        __syncthreads();
        v16h a = load_frag_a(Arow, KDIM, k0, lane);
#pragma unroll
        for (int nt = 0; nt < 4; ++nt) {
            v16h b = load_frag_b(&Bs[nt * 16 * 32], 32, 0, lane);
            acc[nt] = __builtin_amdgcn_wmma_f32_16x16x32_f16(
                false, a, false, b, (short)0, acc[nt], false, false);
        }
    }
    int rr = (lane >> 4) << 3, cc = lane & 15;
#pragma unroll
    for (int nt = 0; nt < 4; ++nt) {
        int n = n0 + nt * 16 + cc;
        float bv = bias[n];
#pragma unroll
        for (int j = 0; j < 8; ++j) {
            int m = m0 + rr + j;
            float v = acc[nt][j] + bv;
            if (TANH) Ch[(size_t)m * NDIM + n] = (_Float16)tanhf(v);
            else      Cf[(size_t)m * NDIM + n] = v;
        }
    }
}

// ---------------- in-place L2 row normalize (fp32) + f16 copy ----------------
__global__ void k_norm(float* __restrict__ z, _Float16* __restrict__ zn16) {
    int lane = threadIdx.x & 31, wave = threadIdx.x >> 5;
    int row = blockIdx.x * 8 + wave;
    float* p = z + (size_t)row * D_;
    float vals[8]; float s = 0.f;
#pragma unroll
    for (int i = 0; i < 8; ++i) { vals[i] = p[i * 32 + lane]; s += vals[i] * vals[i]; }
#pragma unroll
    for (int off = 16; off > 0; off >>= 1) s += __shfl_xor(s, off, 32);
    float inv = 1.0f / fmaxf(sqrtf(s), 1e-12f);
#pragma unroll
    for (int i = 0; i < 8; ++i) {
        float v = vals[i] * inv;
        p[i * 32 + lane] = v;
        zn16[(size_t)row * D_ + i * 32 + lane] = (_Float16)v;
    }
}

// ---------------- WMMA distance + argmin ----------------
// d'_k = ||e_k||^2 - 2 * (z_norm . e_k). 8 waves per block, each wave owns a
// 16-row M tile; the 16-code x 256-d f16 tile is staged in LDS and shared.
__global__ void k_dist(const _Float16* __restrict__ Zn, const _Float16* __restrict__ CB,
                       const float* __restrict__ cbn, int* __restrict__ idx_out,
                       float* __restrict__ idx_out_f) {
    __shared__ __align__(16) _Float16 Cs[16 * 256];
    int tid = threadIdx.x, lane = tid & 31, wave = tid >> 5;
    int m0 = blockIdx.x * 128 + wave * 16;
    const _Float16* Arow = Zn + (size_t)m0 * D_;
    v16h a[8];
#pragma unroll
    for (int kc = 0; kc < 8; ++kc) a[kc] = load_frag_a(Arow, D_, kc * 32, lane);
    float best[8]; int bidx[8];
#pragma unroll
    for (int j = 0; j < 8; ++j) { best[j] = 3.0e38f; bidx[j] = 0; }
    int r = tid >> 4, s = tid & 15;          // 16 rows x 512B, 32B per thread
    for (int n0 = 0; n0 < K_; n0 += 16) {
        __syncthreads();
        const _Float16* src = &CB[(size_t)(n0 + r) * D_ + s * 8];
        *(v8h*)&Cs[r * D_ + s * 8]        = *(const v8h*)(src);
        *(v8h*)&Cs[r * D_ + (s + 16) * 8] = *(const v8h*)(src + 128);
        if (n0 + 16 < K_)                     // global_prefetch next code tile
            __builtin_prefetch(&CB[(size_t)(n0 + 16 + r) * D_ + s * 8], 0, 1);
        __syncthreads();
        v8f c = {};
#pragma unroll
        for (int kc = 0; kc < 8; ++kc) {
            v16h b = load_frag_b(Cs, D_, kc * 32, lane);
            c = __builtin_amdgcn_wmma_f32_16x16x32_f16(
                false, a[kc], false, b, (short)0, c, false, false);
        }
        int code = n0 + (lane & 15);
        float cn = cbn[code];
#pragma unroll
        for (int j = 0; j < 8; ++j) {
            float dv = cn - 2.0f * c[j];
            if (dv < best[j]) { best[j] = dv; bidx[j] = code; }
        }
    }
    int rr = (lane >> 4) << 3, cc = lane & 15;
#pragma unroll
    for (int j = 0; j < 8; ++j) {
        float v = best[j]; int bi = bidx[j];
#pragma unroll
        for (int off = 8; off > 0; off >>= 1) {
            float ov = __shfl_xor(v, off, 16);
            int   oi = __shfl_xor(bi, off, 16);
            if (ov < v || (ov == v && oi < bi)) { v = ov; bi = oi; }  // first-min tie
        }
        if (cc == 0) {
            int m = m0 + rr + j;
            idx_out[m]   = bi;
            idx_out_f[m] = (float)bi;
        }
    }
}

// ---------------- loss: 1.25 * mean((z_norm - l2norm(cb[idx]))^2) ------------
__global__ void k_zero(float* out) { out[0] = 0.f; }

__global__ void k_loss(const float* __restrict__ zn, const float* __restrict__ cb,
                       const int* __restrict__ idx, float* __restrict__ out) {
    __shared__ float red[8];
    int row = blockIdx.x, t = threadIdx.x;
    int lane = t & 31, wave = t >> 5;
    int code = idx[row];
    float v = cb[(size_t)code * D_ + t];
    float s = v * v;
#pragma unroll
    for (int off = 16; off > 0; off >>= 1) s += __shfl_xor(s, off, 32);
    if (lane == 0) red[wave] = s;
    __syncthreads();
    float tot = 0.f;
#pragma unroll
    for (int w = 0; w < 8; ++w) tot += red[w];
    float inv = 1.0f / fmaxf(sqrtf(tot), 1e-12f);
    float d = zn[(size_t)row * D_ + t] - v * inv;
    s = d * d;
    __syncthreads();
#pragma unroll
    for (int off = 16; off > 0; off >>= 1) s += __shfl_xor(s, off, 32);
    if (lane == 0) red[wave] = s;
    __syncthreads();
    if (t == 0) {
        tot = 0.f;
        for (int w = 0; w < 8; ++w) tot += red[w];
        atomicAdd(out, tot * (1.25f / ((float)M_ * (float)D_)));
    }
}

// ---------------- launcher ----------------
extern "C" void kernel_launch(void* const* d_in, const int* in_sizes, int n_in,
                              void* d_out, int out_size, void* d_ws, size_t ws_size,
                              hipStream_t stream) {
    (void)in_sizes; (void)n_in; (void)out_size; (void)ws_size;
    const float* h  = (const float*)d_in[0];
    const float* W1 = (const float*)d_in[1];
    const float* b1 = (const float*)d_in[2];
    const float* W2 = (const float*)d_in[3];
    const float* b2 = (const float*)d_in[4];
    const float* cb = (const float*)d_in[5];
    float* out = (float*)d_out;

    char* p = (char*)d_ws;
    auto carve = [&](size_t bytes) -> char* {
        char* r = p; p += (bytes + 255) & ~(size_t)255; return r;
    };
    _Float16* h16  = (_Float16*)carve((size_t)M_ * E_ * 2);
    _Float16* W1T  = (_Float16*)carve((size_t)E_ * E_ * 2);
    _Float16* W2T  = (_Float16*)carve((size_t)D_ * E_ * 2);
    _Float16* cb16 = (_Float16*)carve((size_t)K_ * D_ * 2);
    float*    cbn  = (float*)   carve((size_t)K_ * 4);
    _Float16* z116 = (_Float16*)carve((size_t)M_ * E_ * 2);
    float*    z    = (float*)   carve((size_t)M_ * D_ * 4);
    _Float16* zn16 = (_Float16*)carve((size_t)M_ * D_ * 2);
    int*      idx  = (int*)     carve((size_t)M_ * 4);

    k_cvt_f16<<<4096, 256, 0, stream>>>(h, h16, M_ * E_);
    k_transpose_f16<<<2304, 256, 0, stream>>>(W1, W1T, E_, E_);
    k_transpose_f16<<<768,  256, 0, stream>>>(W2, W2T, E_, D_);
    k_cbprep<<<K_ / 8, 256, 0, stream>>>(cb, cb16, cbn);

    k_gemm<E_, true ><<<dim3(M_ / 128, E_ / 64), 256, 0, stream>>>(h16,  W1T, b1, z116, nullptr, E_);
    k_gemm<E_, false><<<dim3(M_ / 128, D_ / 64), 256, 0, stream>>>(z116, W2T, b2, nullptr, z,    D_);

    k_norm<<<M_ / 8, 256, 0, stream>>>(z, zn16);

    k_zero<<<1, 1, 0, stream>>>(out);
    k_dist<<<M_ / 128, 256, 0, stream>>>(zn16, cb16, cbn, idx, out + 1);
    k_loss<<<M_, 256, 0, stream>>>(z, cb, idx, out);
}
